// LIFLayer_42030549959278
// MI455X (gfx1250) — matrix operand order repaired and checked by reading, not compiled, not run on hardware
//
#include <hip/hip_runtime.h>

// ---------------------------------------------------------------------------
// Fused LIF layer for MI455X (gfx1250, wave32):
//   phase A: Wx chunk via v_wmma_f32_16x16x32_bf16 (bf16 inputs, f32 accum)
//   phase B: per-column sequential LIF scan, carry held in registers
// ---------------------------------------------------------------------------

typedef __attribute__((ext_vector_type(16))) __bf16 v16bf;
typedef __attribute__((ext_vector_type(4)))  __bf16 v4bf;
typedef __attribute__((ext_vector_type(8)))  float  v8f;

#define B_DIM 64
#define T_DIM 1024
#define I_DIM 512
#define H_DIM 512

#define HTILE      128                 // h columns per workgroup (== blockDim.x)
#define TCHUNK     64                  // timesteps per GEMM/scan pass
#define NTILES     (HTILE / 16)        // 8 WMMA N-tiles per wave
#define LDS_STRIDE 132                 // padded f32 row stride for Wx chunk (bank-safe)
#define WROW       520                 // padded bf16 row stride for W tile (bank-safe)

#define WX_BYTES   (TCHUNK * LDS_STRIDE * 4)           // 33792 B
#define WT_BYTES   (HTILE * WROW * 2)                  // 133120 B
#define SMEM_BYTES (WX_BYTES + WT_BYTES)               // ~163 KB < 320 KB/WGP

__global__ __launch_bounds__(HTILE) void lif_fused_kernel(
    const float* __restrict__ x,      // [B, T, I]
    const float* __restrict__ W,      // [H, I]
    const float* __restrict__ alpha,  // [H]
    const float* __restrict__ u0,     // [B, H]
    const float* __restrict__ s0,     // [B, H]
    float* __restrict__ out)          // [B, T, H]
{
  extern __shared__ char smem[];
  float*  lds_wx = (float*)smem;                   // [TCHUNK][LDS_STRIDE]
  __bf16* lds_w  = (__bf16*)(smem + WX_BYTES);     // [HTILE][WROW]

  const int b     = blockIdx.y;
  const int h0    = blockIdx.x * HTILE;
  const int tid   = threadIdx.x;       // 0..127
  const int wave  = tid >> 5;          // 0..3  -> t-row of 16 within the chunk
  const int lane  = tid & 31;
  const int l15   = lane & 15;
  const int lhalf = lane >> 4;         // 0 or 1

  // ---- one-time: stage W[h0:h0+HTILE, :] into LDS as bf16 (padded rows) ----
  for (int idx = tid * 4; idx < HTILE * I_DIM; idx += HTILE * 4) {
    const int row = idx >> 9;              // / I_DIM
    const int col = idx & (I_DIM - 1);
    const float4 w4 = *(const float4*)(W + ((size_t)(h0 + row) << 9) + col);
    v4bf p;
    p[0] = (__bf16)w4.x; p[1] = (__bf16)w4.y;
    p[2] = (__bf16)w4.z; p[3] = (__bf16)w4.w;
    *(v4bf*)(lds_w + row * WROW + col) = p;
  }

  // ---- scan state: this thread owns output column h0 + tid ----
  const int h = h0 + tid;
  float al = alpha[h];
  al = fminf(fmaxf(al, 0.81873075307798182f), 0.96078943915232320f); // clip(e^-1/5, e^-1/25)
  const float one_m_al = 1.0f - al;
  float u = u0[b * H_DIM + h];
  float s = s0[b * H_DIM + h];

  __syncthreads();

  for (int t0 = 0; t0 < T_DIM; t0 += TCHUNK) {
    // ================= GEMM phase: Wx[t0..t0+63, h0..h0+127] =================
    const int trow = t0 + wave * 16 + l15;                 // A-matrix row (M = lane&15)
    const float* xrow = x + ((size_t)(b * T_DIM + trow) << 9);

    v8f acc[NTILES] = {};

    for (int k0 = 0; k0 < I_DIM; k0 += 32) {
      // A fragment, ISA 16-bit A 16x32 layout:
      //   lanes 0-15 : K = {0..7, 16..23}; lanes 16-31 : +8
      const float* pa = xrow + k0 + lhalf * 8;
      const float4 a0 = *(const float4*)(pa);
      const float4 a1 = *(const float4*)(pa + 4);
      const float4 a2 = *(const float4*)(pa + 16);
      const float4 a3 = *(const float4*)(pa + 20);
      v16bf af;
      af[0]  = (__bf16)a0.x; af[1]  = (__bf16)a0.y;
      af[2]  = (__bf16)a0.z; af[3]  = (__bf16)a0.w;
      af[4]  = (__bf16)a1.x; af[5]  = (__bf16)a1.y;
      af[6]  = (__bf16)a1.z; af[7]  = (__bf16)a1.w;
      af[8]  = (__bf16)a2.x; af[9]  = (__bf16)a2.y;
      af[10] = (__bf16)a2.z; af[11] = (__bf16)a2.w;
      af[12] = (__bf16)a3.x; af[13] = (__bf16)a3.y;
      af[14] = (__bf16)a3.z; af[15] = (__bf16)a3.w;

#pragma unroll
      for (int n = 0; n < NTILES; ++n) {
        // B fragment, ISA 16-bit B 32x16 layout:
        //   N = lane&15, lanes 0-15 hold K={0..7, 16..23}; lanes 16-31 : +8
        union { uint4 q[2]; v16bf v; } bfr;
        const __bf16* pw = lds_w + (n * 16 + l15) * WROW + k0 + lhalf * 8;
        bfr.q[0] = *(const uint4*)(pw);
        bfr.q[1] = *(const uint4*)(pw + 16);

        acc[n] = __builtin_amdgcn_wmma_f32_16x16x32_bf16(
            false, af, false, bfr.v, (short)0, acc[n], false, false);
      }
    }

    // Prefetch next chunk's A rows while we go do the scan.
    if (t0 + TCHUNK < T_DIM)
      __builtin_prefetch(xrow + (size_t)TCHUNK * I_DIM, 0, 1);

    // C layout: VGPR r -> row (r + 8*lhalf) within the wave's 16-row band,
    // col = lane&15 within N-tile n. Scatter to padded LDS chunk.
#pragma unroll
    for (int n = 0; n < NTILES; ++n) {
#pragma unroll
      for (int r = 0; r < 8; ++r) {
        lds_wx[(wave * 16 + lhalf * 8 + r) * LDS_STRIDE + n * 16 + l15] =
            acc[n][r];
      }
    }

    __syncthreads();

    // ================= scan phase: sequential LIF over the chunk =============
    float* orow = out + (((size_t)(b * T_DIM + t0)) << 9) + h;
#pragma unroll 4
    for (int tt = 0; tt < TCHUNK; ++tt) {
      const float wx = lds_wx[tt * LDS_STRIDE + tid];
      u = al * (u - s) + one_m_al * wx;
      s = (u > 1.0f) ? 1.0f : 0.0f;
      orow[(size_t)tt << 9] = s;
    }

    __syncthreads();   // before next chunk overwrites lds_wx
  }
}

extern "C" void kernel_launch(void* const* d_in, const int* in_sizes, int n_in,
                              void* d_out, int out_size, void* d_ws, size_t ws_size,
                              hipStream_t stream) {
  (void)in_sizes; (void)n_in; (void)out_size; (void)d_ws; (void)ws_size;

  const float* x     = (const float*)d_in[0];
  const float* W     = (const float*)d_in[1];
  const float* alpha = (const float*)d_in[2];
  const float* u0    = (const float*)d_in[3];
  const float* s0    = (const float*)d_in[4];
  float*       out   = (float*)d_out;

  // Allow >64KB dynamic LDS (CDNA5 WGP has 320 KB). Host-side attr set is
  // not a stream op, safe under graph capture; deterministic every call.
  (void)hipFuncSetAttribute((const void*)lif_fused_kernel,
                            hipFuncAttributeMaxDynamicSharedMemorySize,
                            SMEM_BYTES);

  dim3 grid(H_DIM / HTILE, B_DIM);   // (4, 64) = 256 workgroups
  dim3 block(HTILE);                 // 128 threads = 4 waves (wave32)
  lif_fused_kernel<<<grid, block, SMEM_BYTES, stream>>>(x, W, alpha, u0, s0, out);
}
